// CriticGN_33930241638933
// MI455X (gfx1250) — compile-verified, matrix operand order checked by reading, not run on hardware
//
#include <hip/hip_runtime.h>
#include <math.h>

#define N_NODES  100000
#define N_EDGES  1600000
#define N_GRAPHS 256
#define F_IN     32
#define F_HID    64
#define F_OUT    64

// Flip to 0 if the inline-asm async-to-LDS syntax is rejected.
#define USE_ASYNC_LDS_ASM 1

typedef __attribute__((ext_vector_type(2)))  float    v2f;
typedef __attribute__((ext_vector_type(8)))  float    v8f;
typedef __attribute__((ext_vector_type(16))) _Float16 v16h;

#if __has_builtin(__builtin_amdgcn_wmma_f32_16x16x4_f32)
#define HAVE_WMMA_F32 1
#else
#define HAVE_WMMA_F32 0
#endif

// ---------------------------------------------------------------------------
// Hardware tanh (CDNA5 V_TANH_F32 transcendental) — one TRANS32 op instead of
// the ~40-VALU branchy libm expansion.
// ---------------------------------------------------------------------------
__device__ __forceinline__ float fast_tanh(float v) {
#if __has_builtin(__builtin_amdgcn_tanhf)
  return __builtin_amdgcn_tanhf(v);
#else
  float r;
  asm("v_tanh_f32 %0, %1" : "=v"(r) : "v"(v));
  return r;
#endif
}

// ---------------------------------------------------------------------------
// 16-byte global -> LDS copy via the gfx1250 async-to-LDS path (ASYNCcnt),
// using inline asm (ISA: GLOBAL_LOAD_ASYNC_TO_LDS_B128, per-lane transfer;
// VDST = LDS byte offset = low 32 bits of the flat shared address).
// ---------------------------------------------------------------------------
__device__ __forceinline__ void copy16_to_lds(const float* __restrict__ g,
                                              float* __restrict__ l) {
#if USE_ASYNC_LDS_ASM
  const unsigned int       lds_off = (unsigned int)(uintptr_t)l;
  const unsigned long long gaddr   = (unsigned long long)(uintptr_t)g;
  asm volatile("global_load_async_to_lds_b128 %0, %1, off"
               :
               : "v"(lds_off), "v"(gaddr)
               : "memory");
#else
  *(float4*)l = *(const float4*)g;
#endif
}

__device__ __forceinline__ void wait_async_lds() {
#if USE_ASYNC_LDS_ASM
  asm volatile("s_wait_asynccnt 0x0" ::: "memory");
#endif
}

// Stage both weight matrices (K x 64 each) into LDS; all threads participate.
template<int K>
__device__ __forceinline__ void stage_weights(const float* __restrict__ Wrel,
                                              const float* __restrict__ Wroot,
                                              float* __restrict__ sWrel,
                                              float* __restrict__ sWroot) {
  constexpr int VECS = (K * 64) / 4;  // float4 chunks per matrix
  for (int i = threadIdx.x; i < VECS; i += blockDim.x) {
    copy16_to_lds(Wrel + i * 4, sWrel + i * 4);
    copy16_to_lds(Wroot + i * 4, sWroot + i * 4);
  }
  wait_async_lds();
  __syncthreads();
}

// ---------------------------------------------------------------------------
// Wave-level 16x64 tile of: C += Arel(16xK) @ Wrel(Kx64) + Aroot(16xK) @ Wroot(Kx64)
// A operands stream from global; B operands come from LDS (ds_load).
// ---------------------------------------------------------------------------
template<int K>
__device__ __forceinline__ void tile_mma(const float* __restrict__ Arel,
                                         const float* __restrict__ Aroot,
                                         const float* __restrict__ Wrel,
                                         const float* __restrict__ Wroot,
                                         int rowBase, int lane, v8f c[4]) {
  const int l16  = lane & 15;
  const int half = lane >> 4;                 // 0: front half-wave, 1: back
  const float* prel  = Arel  + (size_t)(rowBase + l16) * K;
  const float* proot = Aroot + (size_t)(rowBase + l16) * K;
#if HAVE_WMMA_F32
  // V_WMMA_F32_16X16X4_F32: A = 16x4 f32 (lanes0-15 K=k,k+1; lanes16-31 K=k+2,k+3)
  for (int k = 0; k < K; k += 4) {
    const int ka = k + half * 2;
    v2f ar; ar.x = prel[ka];  ar.y = prel[ka + 1];
    v2f ax; ax.x = proot[ka]; ax.y = proot[ka + 1];
#pragma unroll
    for (int t = 0; t < 4; ++t) {
      const int col = t * 16 + l16;
      v2f br; br.x = Wrel[ka * 64 + col];  br.y = Wrel[(ka + 1) * 64 + col];
      v2f bx; bx.x = Wroot[ka * 64 + col]; bx.y = Wroot[(ka + 1) * 64 + col];
      c[t] = __builtin_amdgcn_wmma_f32_16x16x4_f32(false, ar, false, br, (short)0, c[t], false, false);
      c[t] = __builtin_amdgcn_wmma_f32_16x16x4_f32(false, ax, false, bx, (short)0, c[t], false, false);
    }
  }
#else
  // Fallback: fp16 inputs, fp32 accumulate (codegen-confirmed builtin). K % 32 == 0.
  for (int k0 = 0; k0 < K; k0 += 32) {
    v16h ar, ax;
#pragma unroll
    for (int e = 0; e < 16; ++e) {
      const int kk = k0 + (e >> 3) * 16 + half * 8 + (e & 7);
      ar[e] = (_Float16)prel[kk];
      ax[e] = (_Float16)proot[kk];
    }
#pragma unroll
    for (int t = 0; t < 4; ++t) {
      const int col = t * 16 + l16;
      v16h br, bx;
#pragma unroll
      for (int e = 0; e < 16; ++e) {
        const int kk = k0 + (e >> 3) * 16 + half * 8 + (e & 7);
        br[e] = (_Float16)Wrel[kk * 64 + col];
        bx[e] = (_Float16)Wroot[kk * 64 + col];
      }
      c[t] = __builtin_amdgcn_wmma_f32_16x16x32_f16(false, ar, false, br, (short)0, c[t], false, false);
      c[t] = __builtin_amdgcn_wmma_f32_16x16x32_f16(false, ax, false, bx, (short)0, c[t], false, false);
    }
  }
#endif
}

// ---------------------------------------------------------------------------
// Zero workspace (grid-stride)
// ---------------------------------------------------------------------------
__global__ void zero_kernel(float* __restrict__ ws, long long n) {
  long long i = (long long)blockIdx.x * blockDim.x + threadIdx.x;
  const long long stride = (long long)gridDim.x * blockDim.x;
  for (; i < n; i += stride) ws[i] = 0.0f;
}

// ---------------------------------------------------------------------------
// Edge scatter-add: agg[dst] += feat[src].  F/4 threads per edge, float4 loads,
// fp32 global atomics (land in L2 — whole working set fits in 192 MB).
// ---------------------------------------------------------------------------
template<int F>
__global__ void scatter_add_kernel(const float* __restrict__ feat,
                                   const int* __restrict__ ei,
                                   float* __restrict__ agg) {
  constexpr int PARTS = F / 4;
  const long long tid = (long long)blockIdx.x * blockDim.x + threadIdx.x;
  if (tid >= (long long)N_EDGES * PARTS) return;
  const int edge = (int)(tid / PARTS);
  const int part = (int)(tid & (PARTS - 1));
  const int src = ei[edge];
  const int dst = ei[N_EDGES + edge];
  const float4 v = *(const float4*)(feat + (size_t)src * F + part * 4);
  float* p = agg + (size_t)dst * F + part * 4;
  atomicAdd(p + 0, v.x);
  atomicAdd(p + 1, v.y);
  atomicAdd(p + 2, v.z);
  atomicAdd(p + 3, v.w);
}

// ---------------------------------------------------------------------------
// Layer 1: h1 = tanh(agg1 @ W1_rel + b1 + x @ W1_root)     (K=32, out 64)
// ---------------------------------------------------------------------------
__global__ __launch_bounds__(256) void gemm1_kernel(const float* __restrict__ agg1,
                                                    const float* __restrict__ x,
                                                    const float* __restrict__ Wrel,
                                                    const float* __restrict__ Wroot,
                                                    const float* __restrict__ bias,
                                                    float* __restrict__ h1) {
  __shared__ float sWrel[F_IN * 64];
  __shared__ float sWroot[F_IN * 64];
  stage_weights<F_IN>(Wrel, Wroot, sWrel, sWroot);

  const int wave = blockIdx.x * (blockDim.x >> 5) + (threadIdx.x >> 5);
  if (wave < N_NODES / 16) {                       // wave-uniform: EXEC all-ones inside
    const int lane = threadIdx.x & 31;
    const int rowBase = wave * 16;
    v8f c[4] = {};
    tile_mma<F_IN>(agg1, x, sWrel, sWroot, rowBase, lane, c);
    const int l16 = lane & 15, half = lane >> 4;
#pragma unroll
    for (int t = 0; t < 4; ++t) {
      const int col = t * 16 + l16;
      const float b = bias[col];
#pragma unroll
      for (int r = 0; r < 8; ++r) {
        const int row = rowBase + half * 8 + r;
        h1[(size_t)row * 64 + col] = fast_tanh(c[t][r] + b);
      }
    }
  }
}

// ---------------------------------------------------------------------------
// Layer 2 fused with pooling: v = tanh(agg2 @ W2_rel + b2 + h1 @ W2_root);
// psum[batch[row]] += v.  h2 is never materialized.         (K=64, out 64)
// ---------------------------------------------------------------------------
__global__ __launch_bounds__(256) void gemm2_pool_kernel(const float* __restrict__ agg2,
                                                         const float* __restrict__ h1,
                                                         const float* __restrict__ Wrel,
                                                         const float* __restrict__ Wroot,
                                                         const float* __restrict__ bias,
                                                         const int* __restrict__ batch,
                                                         float* __restrict__ psum) {
  __shared__ float sWrel[F_HID * 64];
  __shared__ float sWroot[F_HID * 64];
  stage_weights<F_HID>(Wrel, Wroot, sWrel, sWroot);

  const int wave = blockIdx.x * (blockDim.x >> 5) + (threadIdx.x >> 5);
  if (wave < N_NODES / 16) {
    const int lane = threadIdx.x & 31;
    const int rowBase = wave * 16;
    v8f c[4] = {};
    tile_mma<F_HID>(agg2, h1, sWrel, sWroot, rowBase, lane, c);
    const int l16 = lane & 15, half = lane >> 4;
#pragma unroll
    for (int t = 0; t < 4; ++t) {
      const int col = t * 16 + l16;
      const float b = bias[col];
#pragma unroll
      for (int r = 0; r < 8; ++r) {
        const int row = rowBase + half * 8 + r;
        const int g = batch[row];
        atomicAdd(&psum[(size_t)g * 64 + col], fast_tanh(c[t][r] + b));
      }
    }
  }
}

// ---------------------------------------------------------------------------
// Per-graph node counts
// ---------------------------------------------------------------------------
__global__ void count_kernel(const int* __restrict__ batch, float* __restrict__ pcnt) {
  const int i = blockIdx.x * blockDim.x + threadIdx.x;
  if (i < N_NODES) atomicAdd(&pcnt[batch[i]], 1.0f);
}

// ---------------------------------------------------------------------------
// out = psum / max(count, 1)
// ---------------------------------------------------------------------------
__global__ void finalize_kernel(const float* __restrict__ psum,
                                const float* __restrict__ pcnt,
                                float* __restrict__ out) {
  const int i = blockIdx.x * blockDim.x + threadIdx.x;
  if (i < N_GRAPHS * F_OUT) {
    const int g = i >> 6;
    out[i] = psum[i] / fmaxf(pcnt[g], 1.0f);
  }
}

// ---------------------------------------------------------------------------
extern "C" void kernel_launch(void* const* d_in, const int* in_sizes, int n_in,
                              void* d_out, int out_size, void* d_ws, size_t ws_size,
                              hipStream_t stream) {
  const float* x       = (const float*)d_in[0];
  const int*   ei      = (const int*)d_in[1];
  const int*   batch   = (const int*)d_in[2];
  const float* W1_rel  = (const float*)d_in[3];
  const float* b1      = (const float*)d_in[4];
  const float* W1_root = (const float*)d_in[5];
  const float* W2_rel  = (const float*)d_in[6];
  const float* b2      = (const float*)d_in[7];
  const float* W2_root = (const float*)d_in[8];
  float* out = (float*)d_out;

  float* ws   = (float*)d_ws;
  float* agg1 = ws;                                    // [N, 32]
  float* h1   = agg1 + (size_t)N_NODES * F_IN;         // [N, 64]
  float* agg2 = h1   + (size_t)N_NODES * F_HID;        // [N, 64]
  float* psum = agg2 + (size_t)N_NODES * F_HID;        // [256, 64]
  float* pcnt = psum + (size_t)N_GRAPHS * F_OUT;       // [256]
  const long long total_ws =
      (long long)N_NODES * (F_IN + F_HID + F_HID) + (long long)N_GRAPHS * F_OUT + N_GRAPHS;

  zero_kernel<<<4096, 256, 0, stream>>>(ws, total_ws);

  // Layer 1 aggregate: agg1[dst] += x[src]   (8 threads/edge)
  {
    const long long th = (long long)N_EDGES * (F_IN / 4);
    scatter_add_kernel<F_IN><<<(int)((th + 255) / 256), 256, 0, stream>>>(x, ei, agg1);
  }

  const int tiles  = N_NODES / 16;                     // 6250 exactly
  const int blocks = (tiles + 7) / 8;                  // 8 waves / block
  gemm1_kernel<<<blocks, 256, 0, stream>>>(agg1, x, W1_rel, W1_root, b1, h1);

  // Layer 2 aggregate: agg2[dst] += h1[src]  (16 threads/edge)
  {
    const long long th = (long long)N_EDGES * (F_HID / 4);
    scatter_add_kernel<F_HID><<<(int)((th + 255) / 256), 256, 0, stream>>>(h1, ei, agg2);
  }

  count_kernel<<<(N_NODES + 255) / 256, 256, 0, stream>>>(batch, pcnt);

  gemm2_pool_kernel<<<blocks, 256, 0, stream>>>(agg2, h1, W2_rel, W2_root, b2, batch, psum);

  finalize_kernel<<<(N_GRAPHS * F_OUT + 255) / 256, 256, 0, stream>>>(psum, pcnt, out);
}
